// BackBone_154618823312
// MI455X (gfx1250) — matrix operand order repaired and checked by reading, not compiled, not run on hardware
//
#include <hip/hip_runtime.h>
#include <hip/hip_bf16.h>

// ---------------- problem constants ----------------
constexpr int T  = 30;    // pred steps
constexpr int Bn = 2048;  // batch
constexpr int Dd = 256;   // encoder dim
constexpr int DH = 128;   // context dim

// ---------------- WMMA vector types ----------------
typedef __attribute__((ext_vector_type(16))) __bf16 v16bf;
typedef __attribute__((ext_vector_type(8)))  __bf16 v8bf;
typedef __attribute__((ext_vector_type(8)))  float  v8f;

// ---------------- workspace layout (bytes) ----------------
constexpr size_t OFF_REP  = 0;                                   // Bn*DH  bf16
constexpr size_t OFF_WK   = OFF_REP  + (size_t)Bn * DH * 2;      // T*Dd*DH bf16
constexpr size_t OFF_PRED = OFF_WK   + (size_t)T * Dd * DH * 2;  // T*Bn*Dd bf16
constexpr size_t OFF_ENC  = OFF_PRED + (size_t)T * Bn * Dd * 2;  // T*Bn*Dd bf16
constexpr size_t OFF_TOT  = OFF_ENC  + (size_t)T * Bn * Dd * 2;  // Bn*Bn f32 (transposed last-step logits)
constexpr size_t OFF_LSE  = OFF_TOT  + (size_t)Bn * Bn * 4;      // Bn f32
constexpr size_t OFF_NCE  = OFF_LSE  + (size_t)Bn * 4;           // T*(Bn/16) f32 partials
constexpr size_t OFF_COR  = OFF_NCE  + (size_t)T * (Bn / 16) * 4;// 1 int

__device__ __forceinline__ unsigned short f32_to_bf16_rne(float f) {
    unsigned int u = __float_as_uint(f);
    u += 0x7FFFu + ((u >> 16) & 1u);   // round to nearest even
    return (unsigned short)(u >> 16);
}

// ---------------- phase 0: f32 -> bf16 streaming convert ----------------
__global__ void cvt_f32_bf16(const float* __restrict__ in,
                             unsigned short* __restrict__ out, int n) {
    for (int i = blockIdx.x * blockDim.x + threadIdx.x; i < n;
         i += gridDim.x * blockDim.x)
        out[i] = f32_to_bf16_rne(in[i]);
}

__global__ void zero_correct(int* correct) { *correct = 0; }

// ---------------- phase 1: pred[t] = rep @ Wk[t]^T + b  (bf16 WMMA) ----
// grid = (Bn/16, T), block = 256 (8 waves). Wave w handles column tiles
// n = w*16 and (w+8)*16 of the 16-row output strip.
__global__ __launch_bounds__(256)
void pred_gemm(const unsigned short* __restrict__ rep_bf,
               const unsigned short* __restrict__ wk_bf,
               const float* __restrict__ bias,
               unsigned short* __restrict__ pred_bf) {
    const int t    = blockIdx.y;
    const int m0   = blockIdx.x * 16;
    const int wv   = threadIdx.x >> 5;
    const int lane = threadIdx.x & 31;
    const int hi   = lane >> 4;      // lane half (A: K-phase, B: K-half, D: M-half)
    const int l    = lane & 15;      // row (A) / column (B,D) within tile

    const __bf16* rep = (const __bf16*)rep_bf;
    const __bf16* wkt = (const __bf16*)wk_bf + (size_t)t * Dd * DH;
    unsigned short* pr = pred_bf + (size_t)t * Bn * Dd;

    const __bf16* arow = rep + (size_t)(m0 + l) * DH;

    for (int nt = wv; nt < Dd / 16; nt += 8) {
        const int n0 = nt * 16;
        v8f acc = {};
#pragma unroll
        for (int k0 = 0; k0 < DH; k0 += 32) {
            v16bf a, b;
            // A 16x32: lane row = l, K chunks {k0+hi*8 .. +8} and {k0+16+hi*8 .. +8}
            ((v8bf*)&a)[0] = *(const v8bf*)(arow + k0 + hi * 8);
            ((v8bf*)&a)[1] = *(const v8bf*)(arow + k0 + 16 + hi * 8);
            // B 32x16: lane col = l, K chunks {k0+hi*8 .. +8} and {k0+16+hi*8 .. +8} (row n0+l of Wk)
            ((v8bf*)&b)[0] = *(const v8bf*)(wkt + (size_t)(n0 + l) * DH + k0 + hi * 8);
            ((v8bf*)&b)[1] = *(const v8bf*)(wkt + (size_t)(n0 + l) * DH + k0 + 16 + hi * 8);
            acc = __builtin_amdgcn_wmma_f32_16x16x32_bf16(
                false, a, false, b, (short)0, acc, false, false);
        }
        const float bv = bias[(size_t)t * Dd + n0 + l];
#pragma unroll
        for (int r = 0; r < 8; ++r) {
            const int m = m0 + r + hi * 8;              // D layout: VGPR r -> row
            pr[(size_t)m * Dd + n0 + l] = f32_to_bf16_rne(acc[r] + bv);
        }
    }
}

// ---------------- phase 2: online LSE over totals[t] ----------------
// grid = (Bn/16, T), block = 256 (8 waves). Block owns rows [m0, m0+16) of
// step t; wave w streams columns [w*256, (w+1)*256) in 16 WMMA tiles while
// maintaining flash-style running (max,sum) per row. Diagonal grabbed raw.
// For t == T-1 the raw 16x16 tiles are also stored *transposed* to ws so the
// accuracy pass reads columns coalesced.
__global__ __launch_bounds__(256)
void totals_lse(const unsigned short* __restrict__ enc_bf,
                const unsigned short* __restrict__ pred_bf,
                float* __restrict__ totalsT,
                float* __restrict__ lse_last,
                float* __restrict__ nce_partial) {
    const int t    = blockIdx.y;
    const int m0   = blockIdx.x * 16;
    const int wv   = threadIdx.x >> 5;
    const int lane = threadIdx.x & 31;
    const int hi   = lane >> 4;
    const int l    = lane & 15;
    const bool last = (t == T - 1);

    __shared__ float lds_m[8][16];
    __shared__ float lds_s[8][16];
    __shared__ float lds_diag[16];

    const __bf16* encT  = (const __bf16*)enc_bf  + (size_t)t * Bn * Dd;
    const __bf16* predT = (const __bf16*)pred_bf + (size_t)t * Bn * Dd;

    // Keep the whole 16x256 A strip resident (8 k-steps x v16bf = 64 VGPRs)
    v16bf a[8];
    const __bf16* arow = encT + (size_t)(m0 + l) * Dd;
#pragma unroll
    for (int kk = 0; kk < 8; ++kk) {
        const int k0 = kk * 32;
        ((v8bf*)&a[kk])[0] = *(const v8bf*)(arow + k0 + hi * 8);
        ((v8bf*)&a[kk])[1] = *(const v8bf*)(arow + k0 + 16 + hi * 8);
    }

    float rm[8], rs[8];
#pragma unroll
    for (int r = 0; r < 8; ++r) { rm[r] = -3.0e38f; rs[r] = 0.0f; }

    for (int ct = 0; ct < 16; ++ct) {
        const int c0 = wv * 256 + ct * 16;
        v8f acc = {};
#pragma unroll
        for (int kk = 0; kk < 8; ++kk) {
            v16bf b;
            ((v8bf*)&b)[0] = *(const v8bf*)(predT + (size_t)(c0 + l) * Dd +
                                      kk * 32 + hi * 8);
            ((v8bf*)&b)[1] = *(const v8bf*)(predT + (size_t)(c0 + l) * Dd +
                                      kk * 32 + 16 + hi * 8);
            acc = __builtin_amdgcn_wmma_f32_16x16x32_bf16(
                false, a[kk], false, b, (short)0, acc, false, false);
        }
        // raw diagonal (exactly one writer per row across the block)
#pragma unroll
        for (int r = 0; r < 8; ++r) {
            const int grow = m0 + r + hi * 8;
            if (grow == c0 + l) lds_diag[r + hi * 8] = acc[r];
        }
        if (last) {                      // store transposed: [col][row]
            float* dst = totalsT + (size_t)(c0 + l) * Bn + m0 + hi * 8;
#pragma unroll
            for (int r = 0; r < 8; ++r) dst[r] = acc[r];
        }
        // online softmax update: reduce over 16 lanes of each half (columns)
#pragma unroll
        for (int r = 0; r < 8; ++r) {
            float v = acc[r];
            float tmax = v;
#pragma unroll
            for (int off = 1; off < 16; off <<= 1)
                tmax = fmaxf(tmax, __shfl_xor(tmax, off, 32));
            const float nm = fmaxf(rm[r], tmax);
            float p = __expf(v - nm);
#pragma unroll
            for (int off = 1; off < 16; off <<= 1)
                p += __shfl_xor(p, off, 32);
            rs[r] = rs[r] * __expf(rm[r] - nm) + p;
            rm[r] = nm;
        }
    }

    if (l == 0) {
#pragma unroll
        for (int r = 0; r < 8; ++r) {
            lds_m[wv][hi * 8 + r] = rm[r];
            lds_s[wv][hi * 8 + r] = rs[r];
        }
    }
    __syncthreads();

    if (threadIdx.x < 16) {
        const int row = threadIdx.x;
        float M = -3.0e38f;
#pragma unroll
        for (int w = 0; w < 8; ++w) M = fmaxf(M, lds_m[w][row]);
        float S = 0.0f;
#pragma unroll
        for (int w = 0; w < 8; ++w) S += lds_s[w][row] * __expf(lds_m[w][row] - M);
        const float lse = M + __logf(S);
        if (last) lse_last[m0 + row] = lse;
        float part = lds_diag[row] - lse;
#pragma unroll
        for (int off = 1; off < 16; off <<= 1)
            part += __shfl_xor(part, off, 32);
        if (row == 0)
            nce_partial[(size_t)t * gridDim.x + blockIdx.x] = part;
    }
}

// ---------------- phase 3: accuracy (column argmax of last step) --------
// probs = softmax(totals[-1], axis=1) -> argmax over axis 0 of
// totals[b,c] - lse[b]. One wave per column; coalesced reads of the
// transposed buffer. Ties break to the lower index (matches jnp.argmax).
__global__ __launch_bounds__(256)
void col_argmax(const float* __restrict__ totalsT,
                const float* __restrict__ lse_last,
                int* __restrict__ correct) {
    const int wv   = threadIdx.x >> 5;
    const int lane = threadIdx.x & 31;
    const int c    = blockIdx.x * 8 + wv;
    const float* col = totalsT + (size_t)c * Bn;

    float best = -3.0e38f;
    int   bidx = 0x7FFFFFFF;
    for (int b = lane; b < Bn; b += 32) {
        const float v = col[b] - lse_last[b];
        if (v > best || (v == best && b < bidx)) { best = v; bidx = b; }
    }
#pragma unroll
    for (int off = 16; off; off >>= 1) {
        const float ov = __shfl_xor(best, off, 32);
        const int   oi = __shfl_xor(bidx, off, 32);
        if (ov > best || (ov == best && oi < bidx)) { best = ov; bidx = oi; }
    }
    if (lane == 0 && bidx == c) atomicAdd(correct, 1);
}

// ---------------- phase 4: finalize 4 scalar outputs --------------------
__global__ __launch_bounds__(256)
void finalize(const float* __restrict__ nce_partial,
              const int* __restrict__ correct, float* __restrict__ out) {
    __shared__ float red[256];
    float acc = 0.0f;
    for (int i = threadIdx.x; i < T * (Bn / 16); i += 256)
        acc += nce_partial[i];
    red[threadIdx.x] = acc;
    __syncthreads();
    for (int st = 128; st; st >>= 1) {
        if (threadIdx.x < st) red[threadIdx.x] += red[threadIdx.x + st];
        __syncthreads();
    }
    if (threadIdx.x == 0) {
        out[0] = (float)(*correct) / (float)Bn;                 // accuracy
        out[1] = red[0] / (-1.0f * (float)(Bn * T));            // nce
        out[2] = (float)Bn;                                     // batch
        out[3] = (float)(Bn * T);                               // calc_step
    }
}

// ---------------- host launcher ----------------
extern "C" void kernel_launch(void* const* d_in, const int* in_sizes, int n_in,
                              void* d_out, int out_size, void* d_ws,
                              size_t ws_size, hipStream_t stream) {
    const float* enc = (const float*)d_in[0];   // [T,Bn,Dd]
    const float* rep = (const float*)d_in[1];   // [Bn,DH]
    const float* wkw = (const float*)d_in[2];   // [T,Dd,DH]
    const float* wkb = (const float*)d_in[3];   // [T,Dd]

    char* ws = (char*)d_ws;
    unsigned short* rep_bf  = (unsigned short*)(ws + OFF_REP);
    unsigned short* wk_bf   = (unsigned short*)(ws + OFF_WK);
    unsigned short* pred_bf = (unsigned short*)(ws + OFF_PRED);
    unsigned short* enc_bf  = (unsigned short*)(ws + OFF_ENC);
    float* totalsT  = (float*)(ws + OFF_TOT);
    float* lse_last = (float*)(ws + OFF_LSE);
    float* nce_part = (float*)(ws + OFF_NCE);
    int*   correct  = (int*)(ws + OFF_COR);

    float* out = (float*)d_out;

    // phase 0: bf16 conversions (streaming, trivially memory bound)
    cvt_f32_bf16<<<512, 256, 0, stream>>>(rep, rep_bf, Bn * DH);
    cvt_f32_bf16<<<1024, 256, 0, stream>>>(wkw, wk_bf, T * Dd * DH);
    cvt_f32_bf16<<<4096, 256, 0, stream>>>(enc, enc_bf, T * Bn * Dd);
    zero_correct<<<1, 1, 0, stream>>>(correct);

    // phase 1: pred = rep @ Wk^T + b  (bf16 WMMA, f32 accumulate)
    pred_gemm<<<dim3(Bn / 16, T), 256, 0, stream>>>(rep_bf, wk_bf, wkb, pred_bf);

    // phase 2: flash-style logsumexp + diagonal over totals (the 64 GFLOP op)
    totals_lse<<<dim3(Bn / 16, T), 256, 0, stream>>>(enc_bf, pred_bf, totalsT,
                                                     lse_last, nce_part);

    // phase 3: last-step column argmax -> accuracy counter
    col_argmax<<<Bn / 8, 256, 0, stream>>>(totalsT, lse_last, correct);

    // phase 4: deterministic reduction of partials + scalar outputs
    finalize<<<1, 256, 0, stream>>>(nce_part, correct, out);
}
